// JittorROIHeads_2628519985782
// MI455X (gfx1250) — matrix operand order repaired and checked by reading, not compiled, not run on hardware
//
#include <hip/hip_runtime.h>
#include <hip/hip_bf16.h>
#include <stdint.h>
#include <stddef.h>

// ---------------------------------------------------------------------------
// Types for CDNA5 WMMA (wave32): v16bf = 16xbf16 (8 VGPRs), v8f = 8xf32 acc.
// ---------------------------------------------------------------------------
typedef __attribute__((ext_vector_type(16))) __bf16        v16bf;
typedef __attribute__((ext_vector_type(8)))  float         v8f;
typedef __attribute__((ext_vector_type(4)))  unsigned int  u32x4;
typedef __attribute__((ext_vector_type(2)))  unsigned int  u32x2;
typedef __attribute__((ext_vector_type(4)))  float         f32x4;
typedef __attribute__((ext_vector_type(4)))  int           i32x4;
typedef __attribute__((ext_vector_type(8)))  int           i32x8;

#define NEGV        (-1e10f)
#define BBOX_CLIPV  4.135166556742356f   // log(1000/16)

__device__ __forceinline__ unsigned short f2bf(float f) {
  // round-to-nearest-even f32 -> bf16 (bit pattern)
  unsigned int x = __float_as_uint(f);
  unsigned int r = (x + 0x7FFFu + ((x >> 16) & 1u)) >> 16;
  return (unsigned short)r;
}

struct Frag32 { u32x4 lo, hi; };
__device__ __forceinline__ v16bf make_frag(u32x4 lo, u32x4 hi) {
  Frag32 f{lo, hi};
  return __builtin_bit_cast(v16bf, f);
}

// ---------------------------------------------------------------------------
// TDM: issue one 2D tile load (128 rows x 32 bf16 elems, row stride = K) into
// LDS at byte offset lds_off. Descriptor bitfields per CDNA5 ISA ch.8.
// tensor_dim1 = rows remaining -> hardware zero-fills OOB rows (M/N tails).
// ---------------------------------------------------------------------------
__device__ __forceinline__ void tdm_load_tile(
    unsigned int lds_off, const unsigned short* gptr,
    int rows_rem, int stride_elems)
{
  unsigned long long ga = (unsigned long long)(uintptr_t)gptr;
  u32x4 g0;
  g0.x = 1u;                                  // count=1 (valid descriptor)
  g0.y = lds_off;                             // lds_addr (bytes)
  g0.z = (unsigned int)ga;                    // global_addr[31:0]
  g0.w = ((unsigned int)(ga >> 32) & 0x01FFFFFFu) | (2u << 30); // [56:32] | type=2
  unsigned int d0 = (unsigned int)stride_elems;   // tensor_dim0 (elems)
  unsigned int d1 = (unsigned int)rows_rem;       // tensor_dim1 (rows)
  i32x8 g1;
  g1[0] = (int)(1u << 16);                    // wg_mask=0, data_size=1 (2 bytes)
  g1[1] = (int)((d0 & 0xFFFFu) << 16);        // [31:16] = tensor_dim0[15:0]
  g1[2] = (int)(((d0 >> 16) & 0xFFFFu) | ((d1 & 0xFFFFu) << 16));
  g1[3] = (int)(((d1 >> 16) & 0xFFFFu) | (32u << 16));  // tile_dim0 = 32
  g1[4] = (int)128;                           // tile_dim1 = 128, tile_dim2 = 0
  g1[5] = (int)d0;                            // tensor_dim0_stride[31:0]
  g1[6] = 0;                                  // stride[47:32]=0, dim1_stride lo=0
  g1[7] = 0;
  i32x4 gz = {0, 0, 0, 0};
#if __clang_major__ >= 23
  i32x8 gz8 = {};
  __builtin_amdgcn_tensor_load_to_lds(g0, g1, gz, gz, gz8, 0);
#else
  __builtin_amdgcn_tensor_load_to_lds(g0, g1, gz, gz, 0);
#endif
}

// ---------------------------------------------------------------------------
// GEMM: Out[M,N] = act(A[M,K] @ BT[N,K]^T + bias).
// A: bf16 [M][K] row-major.  BT: bf16 [N][K] row-major (pre-transposed).
// Block tile 128x128, BK=32, 256 threads = 8 waves (4 M x 2 N), each wave
// owns 32x64 = 2x4 WMMA 16x16 tiles.  Tiles staged to LDS by the Tensor Data
// Mover with double buffering; fragments are aligned 16B LDS loads matching
// the ISA bf16 WMMA VGPR layouts:
//   A 16x32: lane L holds row M=L&15; halves 0-7 -> K=(L>=16?8:0)+0..7,
//            halves 8-15 -> K=16+(L>=16?8:0)+0..7.
//   B 32x16: lane L holds col N=L&15; halves 0-7 -> K=(L>=16?8:0)+0..7,
//            halves 8-15 -> K=16+(L>=16?8:0)+0..7.
//   C/D 16x16 f32: vgpr r -> M = r + (L>=16?8:0), N = L&15.
// ---------------------------------------------------------------------------
template <bool RELU, bool OUT_BF16>
__global__ __launch_bounds__(256) void gemm_tdm_wmma(
    const unsigned short* __restrict__ A,
    const unsigned short* __restrict__ BT,
    const float* __restrict__ bias, void* __restrict__ Outv,
    int M, int N, int K, int ldo)
{
  __shared__ unsigned short Abuf[2][128 * 32];
  __shared__ unsigned short Bbuf[2][128 * 32];

  const int tid    = threadIdx.x;
  const int lane   = tid & 31;
  const int wave   = tid >> 5;
  const int wm     = wave & 3;        // wave M index (0..3)
  const int wn     = wave >> 2;       // wave N index (0..1)
  const int laneHi = (lane >> 4) & 1; // lane half
  const int lane15 = lane & 15;

  const int n0 = blockIdx.x * 128;
  const int m0 = blockIdx.y * 128;
  const int nk = K >> 5;              // K always a multiple of 32 here

  const unsigned int aoff0 = (unsigned int)(uintptr_t)&Abuf[0][0];
  const unsigned int aoff1 = (unsigned int)(uintptr_t)&Abuf[1][0];
  const unsigned int boff0 = (unsigned int)(uintptr_t)&Bbuf[0][0];
  const unsigned int boff1 = (unsigned int)(uintptr_t)&Bbuf[1][0];

  v8f acc[2][4] = {};

  // prologue: wave 0 issues the first tile pair
  if (wave == 0) {
    tdm_load_tile(aoff0, A  + (size_t)m0 * K, M - m0, K);
    tdm_load_tile(boff0, BT + (size_t)n0 * K, N - n0, K);
  }

  for (int i = 0; i < nk; ++i) {
    __builtin_amdgcn_s_wait_tensorcnt(0);  // tile i resident (nop for waves 1-7)
    __syncthreads();                        // all waves: data i ready, buf i^1 free
    if (wave == 0 && (i + 1) < nk) {
      unsigned int ao = ((i + 1) & 1) ? aoff1 : aoff0;
      unsigned int bo = ((i + 1) & 1) ? boff1 : boff0;
      tdm_load_tile(ao, A  + (size_t)m0 * K + (i + 1) * 32, M - m0, K);
      tdm_load_tile(bo, BT + (size_t)n0 * K + (i + 1) * 32, N - n0, K);
    }
    const unsigned short* Ab = &Abuf[i & 1][0];
    const unsigned short* Bb = &Bbuf[i & 1][0];

    v16bf afr[2], bfr[4];
#pragma unroll
    for (int mt = 0; mt < 2; ++mt) {
      int mrow = wm * 32 + mt * 16 + lane15;
      const unsigned short* pa = Ab + mrow * 32 + laneHi * 8;
      afr[mt] = make_frag(*(const u32x4*)pa, *(const u32x4*)(pa + 16));
    }
#pragma unroll
    for (int nt = 0; nt < 4; ++nt) {
      int ncol = wn * 64 + nt * 16 + lane15;
      const unsigned short* pb = Bb + ncol * 32 + laneHi * 8;
      bfr[nt] = make_frag(*(const u32x4*)pb, *(const u32x4*)(pb + 16));
    }
#pragma unroll
    for (int mt = 0; mt < 2; ++mt)
#pragma unroll
      for (int nt = 0; nt < 4; ++nt)
        acc[mt][nt] = __builtin_amdgcn_wmma_f32_16x16x32_bf16(
            false, afr[mt], false, bfr[nt], (short)0, acc[mt][nt], false, false);
    __syncthreads();                        // done reading buf i
  }

  // ---- epilogue: bias (+ReLU), store f32 or bf16 ----
#pragma unroll
  for (int mt = 0; mt < 2; ++mt) {
#pragma unroll
    for (int nt = 0; nt < 4; ++nt) {
      int col = n0 + wn * 64 + nt * 16 + lane15;
      if (col >= N) continue;
      float bb = bias[col];
#pragma unroll
      for (int r = 0; r < 8; ++r) {
        int row = m0 + wm * 32 + mt * 16 + laneHi * 8 + r;
        if (row >= M) continue;
        float v = acc[mt][nt][r] + bb;
        if (RELU) v = v > 0.f ? v : 0.f;
        if (OUT_BF16)
          ((unsigned short*)Outv)[(size_t)row * ldo + col] = f2bf(v);
        else
          ((float*)Outv)[(size_t)row * ldo + col] = v;
      }
    }
  }
}

// ---------------------------------------------------------------------------
// Prologue kernels: f32 -> bf16 convert (same layout), and fused
// transpose+convert W[K][N] f32 -> WT[N][K] bf16 (32x32 LDS tiles).
// ---------------------------------------------------------------------------
__global__ void cvt_f32_bf16_kernel(const float* __restrict__ in,
                                    unsigned short* __restrict__ out, size_t n4)
{
  size_t i = (size_t)blockIdx.x * blockDim.x + threadIdx.x;
  if (i >= n4) return;
  f32x4 v = *(const f32x4*)(in + i * 4);
  u32x2 pk;
  pk.x = (unsigned)f2bf(v.x) | ((unsigned)f2bf(v.y) << 16);
  pk.y = (unsigned)f2bf(v.z) | ((unsigned)f2bf(v.w) << 16);
  *(u32x2*)(out + i * 4) = pk;
}

__global__ __launch_bounds__(256) void transpose_cvt_kernel(
    const float* __restrict__ W, unsigned short* __restrict__ WT, int K, int N)
{
  __shared__ float t[32][33];
  int kb = blockIdx.x * 32, nb = blockIdx.y * 32;
  int tx = threadIdx.x, ty = threadIdx.y;  // (32, 8)
#pragma unroll
  for (int r = ty; r < 32; r += 8) {
    int k = kb + r, nn = nb + tx;
    t[r][tx] = (k < K && nn < N) ? W[(size_t)k * N + nn] : 0.f;
  }
  __syncthreads();
#pragma unroll
  for (int r = ty; r < 32; r += 8) {
    int nn = nb + r, k = kb + tx;
    if (nn < N && k < K) WT[(size_t)nn * K + k] = f2bf(t[tx][r]);
  }
}

// ---------------------------------------------------------------------------
// Per-proposal postprocess: softmax over 91 classes, box decode/clip,
// validity mask, class-offset boxes + areas for class-aware NMS.
// ---------------------------------------------------------------------------
__global__ void postprocess_kernel(
    const float* __restrict__ logits, const float* __restrict__ breg,
    const float* __restrict__ props,
    const int* __restrict__ imh, const int* __restrict__ imw,
    float* __restrict__ nmsb, float* __restrict__ boxes,
    float* __restrict__ sm, float* __restrict__ score,
    float* __restrict__ areas, int N, int C)
{
  int n = blockIdx.x * blockDim.x + threadIdx.x;
  if (n >= N) return;
  float Him = (float)(*imh), Wim = (float)(*imw);
  const float* l = logits + (size_t)n * C;
  float mx = l[0];
  for (int c = 1; c < C; ++c) mx = fmaxf(mx, l[c]);
  float ssum = 0.f;
  for (int c = 0; c < C; ++c) ssum += expf(l[c] - mx);
  float inv = 1.f / ssum;

  float x1 = props[n * 4 + 0], y1 = props[n * 4 + 1];
  float x2 = props[n * 4 + 2], y2 = props[n * 4 + 3];
  float pw = x2 - x1, ph = y2 - y1;
  float pcx = x1 + 0.5f * pw, pcy = y1 + 0.5f * ph;
  float offscale = fmaxf(Wim, Him) + 1.f;

  for (int c = 1; c < C; ++c) {
    float sc = expf(l[c] - mx) * inv;
    const float* r = breg + ((size_t)n * C + c) * 4;  // = n*4C + c*4
    float dx = r[0] * 0.1f, dy = r[1] * 0.1f;
    float dw = fminf(r[2] * 0.2f, BBOX_CLIPV);
    float dh = fminf(r[3] * 0.2f, BBOX_CLIPV);
    float cx = dx * pw + pcx, cy = dy * ph + pcy;
    float w = expf(dw) * pw, hh = expf(dh) * ph;
    float bx1 = fminf(fmaxf(cx - 0.5f * w, 0.f), Wim);
    float by1 = fminf(fmaxf(cy - 0.5f * hh, 0.f), Him);
    float bx2 = fminf(fmaxf(cx + 0.5f * w, 0.f), Wim);
    float by2 = fminf(fmaxf(cy + 0.5f * hh, 0.f), Him);
    float bw = bx2 - bx1, bh = by2 - by1;
    bool valid = (sc > 0.05f) && (bw >= 0.01f) && (bh >= 0.01f);
    size_t j = (size_t)n * (C - 1) + (c - 1);
    float off = (float)c * offscale;
    nmsb[j * 4 + 0] = bx1 + off; nmsb[j * 4 + 1] = by1 + off;
    nmsb[j * 4 + 2] = bx2 + off; nmsb[j * 4 + 3] = by2 + off;
    boxes[j * 4 + 0] = bx1; boxes[j * 4 + 1] = by1;
    boxes[j * 4 + 2] = bx2; boxes[j * 4 + 3] = by2;
    score[j] = sc;
    sm[j]    = valid ? sc : NEGV;
    areas[j] = bw * bh;  // offset cancels in w/h
  }
}

// ---------------------------------------------------------------------------
// Greedy class-aware NMS, 100 rounds, single block (sequential dependency).
// Writes d_out directly: boxes[100*4] | scores[100] | labels[100] (as float).
// ---------------------------------------------------------------------------
__global__ __launch_bounds__(1024) void nms_kernel(
    const float* __restrict__ nmsb, const float* __restrict__ areas,
    float* __restrict__ sm, const float* __restrict__ boxes,
    const float* __restrict__ score, float* __restrict__ out, int Mtot)
{
  __shared__ float sv[1024];
  __shared__ int   si[1024];
  __shared__ float bb[5];
  int tid = threadIdx.x;

  for (int it = 0; it < 100; ++it) {
    float best = -3e38f;
    int   bi   = Mtot;
    for (int j = tid; j < Mtot; j += 1024) {
      float v = sm[j];
      if (v > best) { best = v; bi = j; }
    }
    sv[tid] = best; si[tid] = bi;
    __syncthreads();
    for (int stp = 512; stp > 0; stp >>= 1) {
      if (tid < stp) {
        float ov = sv[tid + stp]; int oi = si[tid + stp];
        if (ov > sv[tid] || (ov == sv[tid] && oi < si[tid])) {
          sv[tid] = ov; si[tid] = oi;
        }
      }
      __syncthreads();
    }
    float bv = sv[0]; int bidx = si[0];
    bool valid = bv > NEGV * 0.5f;

    if (tid == 0) {
      if (valid) {
        out[it * 4 + 0] = boxes[bidx * 4 + 0];
        out[it * 4 + 1] = boxes[bidx * 4 + 1];
        out[it * 4 + 2] = boxes[bidx * 4 + 2];
        out[it * 4 + 3] = boxes[bidx * 4 + 3];
        out[400 + it]   = score[bidx];
        out[500 + it]   = (float)(bidx % 90 + 1);
        bb[0] = nmsb[bidx * 4 + 0]; bb[1] = nmsb[bidx * 4 + 1];
        bb[2] = nmsb[bidx * 4 + 2]; bb[3] = nmsb[bidx * 4 + 3];
        bb[4] = areas[bidx];
      } else {
        out[it * 4 + 0] = 0.f; out[it * 4 + 1] = 0.f;
        out[it * 4 + 2] = 0.f; out[it * 4 + 3] = 0.f;
        out[400 + it] = 0.f; out[500 + it] = 0.f;
      }
    }
    __syncthreads();

    if (valid) {
      float X1 = bb[0], Y1 = bb[1], X2 = bb[2], Y2 = bb[3], A = bb[4];
      for (int j = tid; j < Mtot; j += 1024) {
        float xx1 = fmaxf(X1, nmsb[j * 4 + 0]);
        float yy1 = fmaxf(Y1, nmsb[j * 4 + 1]);
        float xx2 = fminf(X2, nmsb[j * 4 + 2]);
        float yy2 = fminf(Y2, nmsb[j * 4 + 3]);
        float inter = fmaxf(xx2 - xx1, 0.f) * fmaxf(yy2 - yy1, 0.f);
        float iou = inter / (A + areas[j] - inter + 1e-9f);
        if (iou > 0.5f || j == bidx) sm[j] = NEGV;
      }
    }
    __syncthreads();
  }
}

// ---------------------------------------------------------------------------
extern "C" void kernel_launch(void* const* d_in, const int* in_sizes, int n_in,
                              void* d_out, int out_size, void* d_ws, size_t ws_size,
                              hipStream_t stream)
{
  const float* X  = (const float*)d_in[0];   // [2000, 12544]
  const float* P  = (const float*)d_in[1];   // [2000, 4]
  const float* W1 = (const float*)d_in[2];   // [12544, 1024]
  const float* b1 = (const float*)d_in[3];   // [1024]
  const float* W2 = (const float*)d_in[4];   // [1024, 1024]
  const float* b2 = (const float*)d_in[5];   // [1024]
  const float* Wc = (const float*)d_in[6];   // [1024, 91]
  const float* bc = (const float*)d_in[7];   // [91]
  const float* Wr = (const float*)d_in[8];   // [1024, 364]
  const float* br = (const float*)d_in[9];   // [364]
  const int*   ih = (const int*)d_in[10];    // image_h
  const int*   iw = (const int*)d_in[11];    // image_w

  const int Np = 2000, Kin = 12544, R = 1024, C = 91;
  const int Mtot = Np * (C - 1);

  char* p = (char*)d_ws;
  auto carve = [&](size_t bytes) -> void* {
    void* r = (void*)p;
    p += (bytes + 255) & ~(size_t)255;
    return r;
  };
  unsigned short* XT  = (unsigned short*)carve((size_t)Np * Kin * 2);
  unsigned short* W1T = (unsigned short*)carve((size_t)Kin * R * 2);
  unsigned short* W2T = (unsigned short*)carve((size_t)R * R * 2);
  unsigned short* WcT = (unsigned short*)carve((size_t)C * R * 2);
  unsigned short* WrT = (unsigned short*)carve((size_t)4 * C * R * 2);
  unsigned short* H1  = (unsigned short*)carve((size_t)Np * R * 2);
  unsigned short* H2  = (unsigned short*)carve((size_t)Np * R * 2);
  float* logits = (float*)carve((size_t)Np * C * 4);
  float* breg   = (float*)carve((size_t)Np * 4 * C * 4);
  float* nmsb   = (float*)carve((size_t)Mtot * 4 * 4);
  float* boxes  = (float*)carve((size_t)Mtot * 4 * 4);
  float* sm     = (float*)carve((size_t)Mtot * 4);
  float* score  = (float*)carve((size_t)Mtot * 4);
  float* areas  = (float*)carve((size_t)Mtot * 4);

  // --- operand conditioning: bf16 convert + weight transpose ---
  {
    size_t n4 = (size_t)Np * Kin / 4;
    cvt_f32_bf16_kernel<<<(unsigned)((n4 + 255) / 256), 256, 0, stream>>>(X, XT, n4);
    dim3 tb(32, 8);
    transpose_cvt_kernel<<<dim3((Kin + 31) / 32, (R + 31) / 32), tb, 0, stream>>>(W1, W1T, Kin, R);
    transpose_cvt_kernel<<<dim3((R + 31) / 32, (R + 31) / 32),  tb, 0, stream>>>(W2, W2T, R, R);
    transpose_cvt_kernel<<<dim3((R + 31) / 32, (C + 31) / 32),  tb, 0, stream>>>(Wc, WcT, R, C);
    transpose_cvt_kernel<<<dim3((R + 31) / 32, (4 * C + 31) / 32), tb, 0, stream>>>(Wr, WrT, R, 4 * C);
  }

  dim3 blk(256);
  // H1 = ReLU(XT @ W1T^T + b1)
  gemm_tdm_wmma<true, true><<<dim3(R / 128, (Np + 127) / 128), blk, 0, stream>>>(
      XT, W1T, b1, H1, Np, R, Kin, R);
  // H2 = ReLU(H1 @ W2T^T + b2)
  gemm_tdm_wmma<true, true><<<dim3(R / 128, (Np + 127) / 128), blk, 0, stream>>>(
      H1, W2T, b2, H2, Np, R, R, R);
  // logits = H2 @ WcT^T + bc
  gemm_tdm_wmma<false, false><<<dim3((C + 127) / 128, (Np + 127) / 128), blk, 0, stream>>>(
      H2, WcT, bc, logits, Np, C, R, C);
  // breg = H2 @ WrT^T + br
  gemm_tdm_wmma<false, false><<<dim3((4 * C + 127) / 128, (Np + 127) / 128), blk, 0, stream>>>(
      H2, WrT, br, breg, Np, 4 * C, R, 4 * C);
  // decode + softmax + masks
  postprocess_kernel<<<(Np + 255) / 256, 256, 0, stream>>>(
      logits, breg, P, ih, iw, nmsb, boxes, sm, score, areas, Np, C);
  // greedy NMS + output gather
  nms_kernel<<<1, 1024, 0, stream>>>(nmsb, areas, sm, boxes, score, (float*)d_out, Mtot);
}
